// MultiHeadSelfCrossAttentionWithEdge_7834020348640
// MI455X (gfx1250) — matrix operand order repaired and checked by reading, not compiled, not run on hardware
//
#include <hip/hip_runtime.h>

#define B_    4
#define N_    256
#define D_    256
#define HALF  128
#define H_    4
#define DH_   32

typedef __bf16 bf16_t;
typedef __attribute__((ext_vector_type(16))) __bf16 v16bf;
typedef __attribute__((ext_vector_type(8)))  float  v8f;
typedef unsigned int v4u __attribute__((ext_vector_type(4)));
typedef int          v8i __attribute__((ext_vector_type(8)));
typedef int          v4i __attribute__((ext_vector_type(4)));

#ifndef __has_builtin
#define __has_builtin(x) 0
#endif
#if __has_builtin(__builtin_amdgcn_tensor_load_to_lds) && \
    __has_builtin(__builtin_amdgcn_s_wait_tensorcnt)
#define USE_TDM 1
#else
#define USE_TDM 0
#endif

// ---------------- CDNA5 WMMA fragment helpers (wave32, contiguous k = 16*(lane/16)+e in A and B) --

// A: 16x32 bf16. lane<16: row=lane, K e0..15 -> 0..15
//               lane>=16: row=lane-16, K e0..15 -> 16..31
__device__ __forceinline__ v16bf load_a_frag_g(const float* __restrict__ A, int lda,
                                               int row0, int k0, int lane) {
  const float* rp = A + (size_t)(row0 + (lane & 15)) * lda + k0 + ((lane >> 4) << 4);
  v16bf a;
#pragma unroll
  for (int e = 0; e < 8; ++e) a[e] = (bf16_t)rp[e];
#pragma unroll
  for (int e = 0; e < 8; ++e) a[e + 8] = (bf16_t)rp[8 + e];
  return a;
}

// B: 32x16 bf16. lane<16: col=lane, K=0..15 ; lane>=16: col=lane-16, K=16..31
__device__ __forceinline__ v16bf load_b_frag_g(const float* __restrict__ W, int ldw,
                                               int k0, int col0, int lane) {
  const float* cp = W + (size_t)(k0 + ((lane >> 4) << 4)) * ldw + col0 + (lane & 15);
  v16bf b;
#pragma unroll
  for (int e = 0; e < 16; ++e) b[e] = (bf16_t)cp[(size_t)e * ldw];
  return b;
}

// ---------------- generic C = relu(A @ W + bias), one wave per 16x16 tile -----
__global__ __launch_bounds__(128)
void gemm_bias_relu_wmma(const float* __restrict__ A, int lda,
                         const float* __restrict__ W, int ldw,
                         const float* __restrict__ bias,
                         float* __restrict__ C, int ldc,
                         int M, int K, int Ncols) {
  const int wid  = threadIdx.x >> 5;
  const int lane = threadIdx.x & 31;
  const int tiles_n = Ncols >> 4;
  const int tile = blockIdx.x * 4 + wid;
  if (tile >= (M >> 4) * tiles_n) return;           // wave-uniform; never taken here
  const int row0 = (tile / tiles_n) << 4;
  const int col0 = (tile % tiles_n) << 4;

  v8f acc = {};
  for (int k0 = 0; k0 < K; k0 += 32) {
    v16bf a = load_a_frag_g(A, lda, row0, k0, lane);
    v16bf b = load_b_frag_g(W, ldw, k0, col0, lane);
    acc = __builtin_amdgcn_wmma_f32_16x16x32_bf16(false, a, false, b,
                                                  (short)0, acc, false, false);
  }
  const int c = col0 + (lane & 15);
  const int rbase = row0 + ((lane >> 4) << 3);
  const float bv = bias[c];
#pragma unroll
  for (int i = 0; i < 8; ++i) {
    float v = acc[i] + bv;
    C[(size_t)(rbase + i) * ldc + c] = v > 0.f ? v : 0.f;
  }
}

// ---------------- pre-swizzle Wk (f32 [256,128]) into bf16 B-fragment order ---
// out layout: [j(col tile)=0..7][kk(k step)=0..7][lane=0..31][e=0..15]
__global__ __launch_bounds__(256)
void arrange_wk(const float* __restrict__ Ws, bf16_t* __restrict__ Ds,
                const float* __restrict__ Wc, bf16_t* __restrict__ Dc) {
  int i = blockIdx.x * 256 + threadIdx.x;           // 32768 total
  int e    = i & 15;
  int lane = (i >> 4) & 31;
  int kk   = (i >> 9) & 7;
  int j    = i >> 12;
  int k   = kk * 32 + ((lane >> 4) << 4) + e;
  int col = (j << 4) + (lane & 15);
  Ds[i] = (bf16_t)Ws[k * HALF + col];
  Dc[i] = (bf16_t)Wc[k * HALF + col];
}

// ---------------- TDM: async-load one 16x256 fp32 tile into LDS ---------------
// D# per CDNA5 ISA. 2D tile, data_size=4B,
// pad_enable: +12 dwords after every 256 dwords -> LDS row stride 268 dwords
// (268 mod 64 banks = 12 -> conflict-free A-fragment reads).
#define STG_ROW 268                       // dwords per staged row
#define STG     (16 * STG_ROW)            // dwords per buffer

#if USE_TDM
__device__ __forceinline__ void tdm_load_tile(unsigned lds_off,
                                              unsigned long long gaddr) {
  v4u g0;
  g0[0] = 1u;                                              // count=1 (valid user D#)
  g0[1] = lds_off;                                         // lds_addr (bytes)
  g0[2] = (unsigned)(gaddr & 0xFFFFFFFFull);               // global_addr[31:0]
  g0[3] = (unsigned)((gaddr >> 32) & 0x01FFFFFFull)        // global_addr[56:32]
          | (2u << 30);                                    // type = 2 ("image")
  v8i g1;
  g1[0] = (int)((2u << 16)      // data_size = 4B
              | (1u << 20)      // pad_enable
              | (7u << 22)      // pad_interval: 256 dwords
              | (11u << 25));   // pad_amount: 12 dwords
  g1[1] = (int)(256u << 16);    // tensor_dim0 = 256 ([79:48] lo16)
  g1[2] = (int)(16u  << 16);    // tensor_dim0 hi=0 | tensor_dim1 = 16 (lo16)
  g1[3] = (int)(256u << 16);    // tensor_dim1 hi=0 | tile_dim0 = 256
  g1[4] = 16;                   // tile_dim1 = 16, tile_dim2 = 0
  g1[5] = 256;                  // tensor_dim0_stride = 256 elements
  g1[6] = 0;
  g1[7] = 0;
  v4i z = {0, 0, 0, 0};
#if __clang_major__ >= 23
  v8i z8 = {0, 0, 0, 0, 0, 0, 0, 0};
  __builtin_amdgcn_tensor_load_to_lds(g0, g1, z, z, z8, 0);
#else
  __builtin_amdgcn_tensor_load_to_lds(g0, g1, z, z, 0);
#endif
}
#endif

// ---------------- fused edge projection + scores + softmax + attn@V ----------
#define EK_STRIDE 132   // f32 16x128 projected tile (padded)

__global__ __launch_bounds__(256)
void edge_attn_fused(const float* __restrict__ edges_s, const float* __restrict__ edges_c,
                     const bf16_t* __restrict__ wk_s,   const bf16_t* __restrict__ wk_c,
                     const float* __restrict__ bk_s,    const float* __restrict__ bk_c,
                     const float* __restrict__ q_s,     const float* __restrict__ q_c,
                     const int*   __restrict__ mask_s,  const int*   __restrict__ mask_c,
                     float* __restrict__ y_cat) {
  __shared__ float  stage[2 * STG];      // double-buffered fp32 edge tiles (TDM dest)
  __shared__ float  ek_lds[16 * EK_STRIDE];
  __shared__ float  sc_lds[H_ * N_];
  __shared__ float  q_lds[HALF];
  __shared__ float  bias_lds[HALF];

  const int tid  = threadIdx.x;
  const int lane = tid & 31;
  const int wid  = tid >> 5;

  int idx = blockIdx.x;
  const int type = idx & 1; idx >>= 1;
  const int b = idx >> 8;
  const int n = idx & 255;

  const float*  edges   = type ? edges_c : edges_s;
  const bf16_t* wk      = type ? wk_c    : wk_s;
  const float*  bk      = type ? bk_c    : bk_s;
  const float*  q       = (type ? q_c : q_s) + (size_t)(b * N_ + n) * HALF;
  const int*    maskrow = (type ? mask_c : mask_s) + (size_t)(b * N_ + n) * N_;

  const size_t ebase = (size_t)(b * N_ + n) * N_ * D_;

#if USE_TDM
  // kick off tile 0 as early as possible (wave 0 only; TDM ignores EXEC)
  if (wid == 0)
    tdm_load_tile((unsigned)(size_t)(void*)&stage[0],
                  (unsigned long long)(size_t)(edges + ebase));
#endif

  // B fragments for this wave's 16 output columns: resident in VGPRs whole block
  v16bf bfrag[8];
  {
    const v16bf* wfr = (const v16bf*)wk + (size_t)wid * 8 * 32;
#pragma unroll
    for (int kk = 0; kk < 8; ++kk) bfrag[kk] = wfr[kk * 32 + lane];
  }
  if (tid < HALF) { q_lds[tid] = q[tid]; bias_lds[tid] = bk[tid]; }
  __syncthreads();

  const float scale = 0.17677669529663687f;  // 1/sqrt(32)
  const int col0 = wid << 4;
  int buf = 0;

  for (int m0 = 0; m0 < N_; m0 += 16) {
#if USE_TDM
    if (wid == 0) {
      if (m0 + 16 < N_) {        // prefetch next tile into other buffer, then
        tdm_load_tile((unsigned)(size_t)(void*)&stage[(buf ^ 1) * STG],
                      (unsigned long long)(size_t)(edges + ebase + (size_t)(m0 + 16) * D_));
        __builtin_amdgcn_s_wait_tensorcnt(1);   // current tile done (in-order)
      } else {
        __builtin_amdgcn_s_wait_tensorcnt(0);
      }
    }
    __syncthreads();             // publish TDM-written LDS to all waves
#else
    for (int i = tid; i < 16 * 256; i += 256) {
      int r = i >> 8, c = i & 255;
      stage[buf * STG + r * STG_ROW + c] = edges[ebase + (size_t)(m0 + r) * D_ + c];
    }
    __syncthreads();
#endif

    // projected-key tile: each wave a 16x16 slab, K=256 in 8 WMMAs
    const float* sb = stage + buf * STG;
    v8f acc = {};
#pragma unroll
    for (int kk = 0; kk < 8; ++kk) {
      const float* ap = sb + (lane & 15) * STG_ROW + kk * 32 + ((lane >> 4) << 4);
      v16bf a;
#pragma unroll
      for (int e = 0; e < 8; ++e) a[e] = (bf16_t)ap[e];
#pragma unroll
      for (int e = 0; e < 8; ++e) a[e + 8] = (bf16_t)ap[8 + e];
      acc = __builtin_amdgcn_wmma_f32_16x16x32_bf16(false, a, false, bfrag[kk],
                                                    (short)0, acc, false, false);
    }
    {
      const int c = col0 + (lane & 15);
      const int rbase = (lane >> 4) << 3;
      const float bv = bias_lds[c];
#pragma unroll
      for (int i = 0; i < 8; ++i) {
        float v = acc[i] + bv;
        ek_lds[(rbase + i) * EK_STRIDE + c] = v > 0.f ? v : 0.f;   // relu
      }
    }
    __syncthreads();

    // scores for these 16 keys: 64 threads = (head, key-row)
    if (tid < 64) {
      int h = tid >> 4, r = tid & 15;
      float s = 0.f;
#pragma unroll
      for (int d = 0; d < DH_; ++d)
        s += ek_lds[r * EK_STRIDE + h * DH_ + d] * q_lds[h * DH_ + d];
      sc_lds[h * N_ + m0 + r] = (maskrow[m0 + r] == 0) ? -1e9f : s * scale;
    }
    __syncthreads();             // also fences stage[] reuse for next iteration
    buf ^= 1;
  }

  // softmax: wave per head
  if (wid < H_) {
    float* sh = sc_lds + wid * N_;
    float mx = -3.4e38f;
    for (int m = lane; m < N_; m += 32) mx = fmaxf(mx, sh[m]);
#pragma unroll
    for (int off = 16; off > 0; off >>= 1) mx = fmaxf(mx, __shfl_xor(mx, off, 32));
    float sum = 0.f;
    for (int m = lane; m < N_; m += 32) { float e = __expf(sh[m] - mx); sh[m] = e; sum += e; }
#pragma unroll
    for (int off = 16; off > 0; off >>= 1) sum += __shfl_xor(sum, off, 32);
    float inv = 1.f / sum;
    for (int m = lane; m < N_; m += 32) sh[m] *= inv;
  }
  __syncthreads();

  // out[h*32+d] = sum_m attn[h,m] * V[b,m,h*32+d]; V lives in y_cat cols 256+type*128
  const float* vbuf = y_cat + (size_t)b * N_ * 512 + 256 + type * HALF;
  float* yout = y_cat + (size_t)(b * N_ + n) * 512 + type * HALF;
  if (tid < HALF) {
    const float* att = sc_lds + (tid >> 5) * N_;
    float o = 0.f;
    for (int m = 0; m < N_; ++m) o += att[m] * vbuf[(size_t)m * 512 + tid];
    yout[tid] = o;
  }
}

// -----------------------------------------------------------------------------
extern "C" void kernel_launch(void* const* d_in, const int* in_sizes, int n_in,
                              void* d_out, int out_size, void* d_ws, size_t ws_size,
                              hipStream_t stream) {
  const float* f1    = (const float*)d_in[0];
  const float* f2    = (const float*)d_in[1];
  const float* edges = (const float*)d_in[2];
  const float* edgec = (const float*)d_in[3];
  const int*   masks = (const int*)d_in[4];
  const int*   maskc = (const int*)d_in[5];
  const float* Wqs = (const float*)d_in[6];  const float* bqs = (const float*)d_in[7];
  const float* Wks = (const float*)d_in[8];  const float* bks = (const float*)d_in[9];
  const float* Wvs = (const float*)d_in[10]; const float* bvs = (const float*)d_in[11];
  const float* Wqc = (const float*)d_in[12]; const float* bqc = (const float*)d_in[13];
  const float* Wkc = (const float*)d_in[14]; const float* bkc = (const float*)d_in[15];
  const float* Wvc = (const float*)d_in[16]; const float* bvc = (const float*)d_in[17];
  const float* Wo  = (const float*)d_in[18]; const float* bo  = (const float*)d_in[19];
  float* out = (float*)d_out;

  char* ws = (char*)d_ws;
  float*  q_s   = (float*)(ws);                       // 1024*128 f32 = 512 KB
  float*  q_c   = (float*)(ws + (512u << 10));        // 512 KB
  float*  y_cat = (float*)(ws + (1024u << 10));       // 1024*512 f32 = 2 MB
  bf16_t* wkA_s = (bf16_t*)(ws + (3072u << 10));      // 64 KB bf16, fragment order
  bf16_t* wkA_c = (bf16_t*)(ws + (3136u << 10));      // 64 KB

  // 0) pre-swizzle edge-key weights to bf16 WMMA B-fragment order
  arrange_wk<<<128, 256, 0, stream>>>(Wks, wkA_s, Wkc, wkA_c);

  // 1) small projections: [1024,256]x[256,128] each (512 tiles -> 128 blocks x 4 waves)
  gemm_bias_relu_wmma<<<128, 128, 0, stream>>>(f1, D_, Wqs, HALF, bqs, q_s,        HALF, 1024, D_, HALF);
  gemm_bias_relu_wmma<<<128, 128, 0, stream>>>(f1, D_, Wvs, HALF, bvs, y_cat + 256, 512, 1024, D_, HALF); // vs_m slot
  gemm_bias_relu_wmma<<<128, 128, 0, stream>>>(f1, D_, Wqc, HALF, bqc, q_c,        HALF, 1024, D_, HALF);
  gemm_bias_relu_wmma<<<128, 128, 0, stream>>>(f2, D_, Wvc, HALF, bvc, y_cat + 384, 512, 1024, D_, HALF); // vc_m slot

  // 2) fused edge projection + attention (self & cross interleaved), one WG per (b,n,type)
  edge_attn_fused<<<2 * B_ * N_, 256, 0, stream>>>(edges, edgec, wkA_s, wkA_c,
                                                   bks, bkc, q_s, q_c, masks, maskc, y_cat);

  // 3) out = relu(y_cat @ Wo + bo): [1024,512]x[512,256] (1024 tiles -> 256 blocks)
  gemm_bias_relu_wmma<<<256, 128, 0, stream>>>(y_cat, 512, Wo, D_, bo, out, D_, 1024, 512, D_);
}
